// SingleAttentionHead_47210280517803
// MI455X (gfx1250) — matrix operand order repaired and checked by reading, not compiled, not run on hardware
//
#include <hip/hip_runtime.h>
#include <hip/hip_bf16.h>

// ---------------------------------------------------------------------------
// SingleAttentionHead for MI455X (gfx1250), bf16 WMMA pipeline + async LDS.
//   x:[8,2048,1024] f32, Wq/Wk/Wv:[1024,64] f32 -> out:[8,2048,64] f32
// ---------------------------------------------------------------------------

typedef __attribute__((ext_vector_type(16))) __bf16 bf16x16;
typedef __attribute__((ext_vector_type(8)))  float  f32x8;
typedef int v4i __attribute__((vector_size(16)));   // b128 payload type

#define C_DIM 1024
#define HDIM  64
#define TSEQ  2048
#define NBATCH 8
#define NROWS (NBATCH * TSEQ)   // 16384

#if defined(__has_builtin)
#if __has_builtin(__builtin_amdgcn_global_load_async_to_lds_b128) && \
    __has_builtin(__builtin_amdgcn_s_wait_asynccnt)
#define USE_ASYNC_COPY 1
#endif
#endif

#ifdef USE_ASYNC_COPY
typedef __attribute__((address_space(1))) v4i* gv4i_p;
typedef __attribute__((address_space(3))) v4i* lv4i_p;
__device__ __forceinline__ void async_b128(const void* g, void* l) {
    __builtin_amdgcn_global_load_async_to_lds_b128((gv4i_p)g, (lv4i_p)l, 0, 0);
}
#endif

__device__ __forceinline__ f32x8 zero8() {
    f32x8 z;
#pragma unroll
    for (int i = 0; i < 8; ++i) z[i] = 0.0f;
    return z;
}

// ---------------------------------------------------------------------------
// Kernel 0: convert + transpose weights: W[C][H] f32 -> Wt[3][H][C] bf16
// ---------------------------------------------------------------------------
__global__ __launch_bounds__(256) void wprep_kernel(
    const float* __restrict__ Wq, const float* __restrict__ Wk,
    const float* __restrict__ Wv, __bf16* __restrict__ Wt)
{
    int idx = blockIdx.x * 256 + threadIdx.x;        // 0 .. 3*65536-1
    int w   = idx >> 16;
    int r   = idx & 0xFFFF;
    int c   = r >> 6;        // 0..1023
    int h   = r & 63;        // 0..63
    const float* W = (w == 0) ? Wq : ((w == 1) ? Wk : Wv);
    Wt[((size_t)w * HDIM + h) * C_DIM + c] = (__bf16)W[(size_t)c * HDIM + h];
}

// ---------------------------------------------------------------------------
// Kernel 1: QKV projection.  Each block: 128 rows x 64 cols of q,k,v.
// 8 waves; wave w owns rows [16w,16w+16).  K-loop in chunks of 32 (bf16 WMMA).
// ---------------------------------------------------------------------------
__global__ __launch_bounds__(256) void qkv_proj_kernel(
    const float* __restrict__ x,      // [16384][1024]
    const __bf16* __restrict__ Wt,    // [3][64][1024] bf16
    __bf16* __restrict__ qkv)         // [3][16384][64] bf16
{
    __shared__ float sx[128][36];     // stride 36 -> 144B rows, 16B aligned

    const int row0 = blockIdx.x * 128;
    const int tid  = threadIdx.x;
    const int wave = tid >> 5;
    const int lane = tid & 31;
    const int hi   = lane >> 4;       // half-wave select
    const int lm   = lane & 15;

    f32x8 acc[3][4];
#pragma unroll
    for (int w = 0; w < 3; ++w)
#pragma unroll
        for (int nt = 0; nt < 4; ++nt) acc[w][nt] = zero8();

    for (int kc = 0; kc < C_DIM; kc += 32) {
        // ---- stage x tile [128][32] fp32 into LDS ----
#ifdef USE_ASYNC_COPY
#pragma unroll
        for (int p = 0; p < 4; ++p) {
            const int r = (tid >> 3) + p * 32;
            const int c = (tid & 7) * 4;
            async_b128(x + (size_t)(row0 + r) * C_DIM + kc + c, &sx[r][c]);
        }
        __builtin_amdgcn_s_wait_asynccnt(0);
#else
#pragma unroll
        for (int p = 0; p < 4; ++p) {
            const int r = (tid >> 3) + p * 32;
            const int c = (tid & 7) * 4;
            const float4 vv =
                *(const float4*)(x + (size_t)(row0 + r) * C_DIM + kc + c);
            *(float4*)&sx[r][c] = vv;
        }
#endif
        if (kc + 32 < C_DIM) {
            // speculative prefetch of next x tile (global_prefetch_b8)
            __builtin_prefetch(x + (size_t)(row0 + (tid >> 3)) * C_DIM + kc + 32, 0, 0);
        }
        __syncthreads();

        // ---- A fragment (16x32 bf16): lanes 0-15 K{0..15}, 16-31 K{16..31}
        bf16x16 a;
        {
            const int m  = 16 * wave + lm;
            const int kb = 16 * hi;
#pragma unroll
            for (int i = 0; i < 8; ++i) a[i]     = (__bf16)sx[m][kb + i];
#pragma unroll
            for (int i = 0; i < 8; ++i) a[8 + i] = (__bf16)sx[m][8 + kb + i];
        }

        // ---- B fragments (32x16 bf16): lane n holds col n, K contiguous ----
#pragma unroll
        for (int w = 0; w < 3; ++w) {
#pragma unroll
            for (int nt = 0; nt < 4; ++nt) {
                const int col = nt * 16 + lm;
                const bf16x16 b = *(const bf16x16*)(
                    Wt + ((size_t)w * HDIM + col) * C_DIM + kc + 16 * hi);
                acc[w][nt] = __builtin_amdgcn_wmma_f32_16x16x32_bf16(
                    false, a, false, b, (short)0, acc[w][nt], false, false);
            }
        }
        __syncthreads();
    }

    // ---- store bf16 q,k,v.  C layout: row = r + 8*hi, col = lm (per N-tile)
#pragma unroll
    for (int w = 0; w < 3; ++w)
#pragma unroll
        for (int nt = 0; nt < 4; ++nt)
#pragma unroll
            for (int r = 0; r < 8; ++r) {
                const int row = row0 + 16 * wave + r + 8 * hi;
                const int col = nt * 16 + lm;
                qkv[((size_t)w * NROWS + row) * HDIM + col] = (__bf16)acc[w][nt][r];
            }
}

// ---------------------------------------------------------------------------
// Kernel 2: causal flash attention over bf16 q,k,v -> fp32 out.
// Block = 128 threads (4 waves) handles one (batch, 64-query tile).
// Wave owns 16 query rows.  Key loop in steps of 32 (two 16-wide N-tiles).
// V tile staged row-major via async copy, overlapped with the S WMMAs.
// ---------------------------------------------------------------------------
__global__ __launch_bounds__(128) void attn_kernel(
    const __bf16* __restrict__ q, const __bf16* __restrict__ k,
    const __bf16* __restrict__ v, float* __restrict__ out)
{
    __shared__ __bf16 sV[32][72];           // V tile row-major [key][dim], 144B rows
    __shared__ __bf16 sP[4][16][32];        // per-wave P tile (row-major)

    const int b    = blockIdx.x >> 5;       // 32 query tiles per batch
    const int qt   = blockIdx.x & 31;
    const int q0   = qt * 64;
    const size_t base = (size_t)b * TSEQ * HDIM;

    const int tid  = threadIdx.x;
    const int wave = tid >> 5;
    const int lane = tid & 31;
    const int hi   = lane >> 4;
    const int lm   = lane & 15;

    // ---- Q fragments (A layout), head dim split into 2 chunks of 32 ----
    bf16x16 aq[2];
    {
        const int mrow_q = q0 + 16 * wave + lm;
        const __bf16* qp = q + base + (size_t)mrow_q * HDIM;
#pragma unroll
        for (int c = 0; c < 2; ++c) {
#pragma unroll
            for (int i = 0; i < 8; ++i) {
                aq[c][i]     = qp[32 * c + 16 * hi + i];
                aq[c][8 + i] = qp[32 * c + 16 * hi + 8 + i];
            }
        }
    }

    f32x8 acc[4];
#pragma unroll
    for (int nt = 0; nt < 4; ++nt) acc[nt] = zero8();
    float mrow[8], lrow[8];
#pragma unroll
    for (int r = 0; r < 8; ++r) { mrow[r] = -3.0e38f; lrow[r] = 0.0f; }

    const float scale = 0.03125f;           // C^-0.5 = 1/32
    const int kend = q0 + 64;               // causal: keys < q0+64

    for (int kt = 0; kt < kend; kt += 32) {
        __syncthreads();                    // previous step done with sV

        // ---- stage V tile [32 keys][64 dims] row-major into LDS ----
        {
            const int key = tid >> 2;           // 0..31
            const int d0  = (tid & 3) * 16;     // 0,16,32,48
            const __bf16* src = v + base + (size_t)(kt + key) * HDIM + d0;
#ifdef USE_ASYNC_COPY
            async_b128(src,     &sV[key][d0]);
            async_b128(src + 8, &sV[key][d0 + 8]);
#else
#pragma unroll
            for (int i = 0; i < 16; ++i) sV[key][d0 + i] = src[i];
#endif
        }

        // ---- S = Q @ K^T  (two 16-wide key N-tiles, K-dim = 64 head dims)
        //      overlaps the async V staging above
        f32x8 s[2];
        s[0] = zero8(); s[1] = zero8();
#pragma unroll
        for (int nt = 0; nt < 2; ++nt) {
#pragma unroll
            for (int c = 0; c < 2; ++c) {
                const bf16x16 bk = *(const bf16x16*)(
                    k + base + (size_t)(kt + nt * 16 + lm) * HDIM + 32 * c + 16 * hi);
                s[nt] = __builtin_amdgcn_wmma_f32_16x16x32_bf16(
                    false, aq[c], false, bk, (short)0, s[nt], false, false);
            }
        }

        // ---- scale, causal mask, online softmax; P -> LDS (bf16) ----
#pragma unroll
        for (int r = 0; r < 8; ++r) {
            const int row_g = q0 + 16 * wave + r + 8 * hi;
            float s0 = s[0][r] * scale;
            float s1 = s[1][r] * scale;
            if (kt + lm      > row_g) s0 = -3.0e38f;
            if (kt + 16 + lm > row_g) s1 = -3.0e38f;

            float mx = fmaxf(s0, s1);
#pragma unroll
            for (int off = 1; off < 16; off <<= 1)
                mx = fmaxf(mx, __shfl_xor(mx, off, 16));
            const float mnew = fmaxf(mrow[r], mx);
            const float p0 = __expf(s0 - mnew);
            const float p1 = __expf(s1 - mnew);
            const float corr = __expf(mrow[r] - mnew);
            mrow[r] = mnew;
            float rs = p0 + p1;
#pragma unroll
            for (int off = 1; off < 16; off <<= 1)
                rs += __shfl_xor(rs, off, 16);
            lrow[r] = lrow[r] * corr + rs;
#pragma unroll
            for (int nt = 0; nt < 4; ++nt) acc[nt][r] *= corr;

            sP[wave][r + 8 * hi][lm]      = (__bf16)p0;
            sP[wave][r + 8 * hi][16 + lm] = (__bf16)p1;
        }
#ifdef USE_ASYNC_COPY
        __builtin_amdgcn_s_wait_asynccnt(0);
#endif
        __syncthreads();                    // sV staged + sP written

        // ---- O += P @ V : A = P (16x32), B = V slices from LDS ----
        bf16x16 ap;
#pragma unroll
        for (int i = 0; i < 8; ++i) {
            ap[i]     = sP[wave][lm][16 * hi + i];
            ap[8 + i] = sP[wave][lm][16 * hi + 8 + i];
        }
#pragma unroll
        for (int nt = 0; nt < 4; ++nt) {
            bf16x16 bv;
            const int dim = nt * 16 + lm;
#pragma unroll
            for (int i = 0; i < 16; ++i) bv[i] = sV[16 * hi + i][dim];
            acc[nt] = __builtin_amdgcn_wmma_f32_16x16x32_bf16(
                false, ap, false, bv, (short)0, acc[nt], false, false);
        }
    }

    // ---- normalize and store fp32 output ----
#pragma unroll
    for (int nt = 0; nt < 4; ++nt)
#pragma unroll
        for (int r = 0; r < 8; ++r) {
            const int row = q0 + 16 * wave + r + 8 * hi;
            const int col = nt * 16 + lm;
            out[base + (size_t)row * HDIM + col] = acc[nt][r] / lrow[r];
        }
}

// ---------------------------------------------------------------------------
// Launch
// ---------------------------------------------------------------------------
extern "C" void kernel_launch(void* const* d_in, const int* in_sizes, int n_in,
                              void* d_out, int out_size, void* d_ws, size_t ws_size,
                              hipStream_t stream) {
    const float* x  = (const float*)d_in[0];
    const float* Wq = (const float*)d_in[1];
    const float* Wk = (const float*)d_in[2];
    const float* Wv = (const float*)d_in[3];
    float* out = (float*)d_out;

    // workspace layout (bf16 elements):
    //   Wt  : 3*64*1024            = 196608
    //   qkv : 3*16384*64           = 3145728
    __bf16* Wt  = (__bf16*)d_ws;
    __bf16* qkv = Wt + (size_t)3 * HDIM * C_DIM;
    __bf16* qb  = qkv;
    __bf16* kb  = qkv + (size_t)NROWS * HDIM;
    __bf16* vb  = qkv + (size_t)2 * NROWS * HDIM;

    wprep_kernel<<<768, 256, 0, stream>>>(Wq, Wk, Wv, Wt);
    qkv_proj_kernel<<<NROWS / 128, 256, 0, stream>>>(x, Wt, qkv);
    attn_kernel<<<NBATCH * (TSEQ / 64), 128, 0, stream>>>(qb, kb, vb, out);
}
